// GlobalFeatureExtractor_44298292691606
// MI455X (gfx1250) — matrix operand order repaired and checked by reading, not compiled, not run on hardware
//
#include <hip/hip_runtime.h>
#include <hip/hip_bf16.h>

typedef __attribute__((ext_vector_type(16))) _Float16 v16h;
typedef __attribute__((ext_vector_type(8)))  _Float16 v8h;
typedef __attribute__((ext_vector_type(8)))  float    v8f;
typedef __attribute__((ext_vector_type(4)))  int      v4i;

// ---------------- CDNA5 async-copy feature detection ----------------
#if defined(__has_builtin)
#  if __has_builtin(__builtin_amdgcn_global_load_async_to_lds_b128) && \
      __has_builtin(__builtin_amdgcn_s_wait_asynccnt)
#    define USE_ASYNC_LDS 1
#  endif
#endif
#ifndef USE_ASYNC_LDS
#  define USE_ASYNC_LDS 0
#endif

#define AS1 __attribute__((address_space(1)))
#define AS3 __attribute__((address_space(3)))

#if USE_ASYNC_LDS
__device__ __forceinline__ void async_cp16(const _Float16* g, _Float16* l) {
  __builtin_amdgcn_global_load_async_to_lds_b128(
      (AS1 v4i*)(AS1 void*)(void*)const_cast<_Float16*>(g),
      (AS3 v4i*)(AS3 void*)l, 0, 0);
}
#endif

// ---------------- problem constants ----------------
#define BB    32
#define HH    56
#define WW_   56
#define CC    192
#define WS_   7
#define SHIFT_ 3
#define NH_   6
#define HD_   32
#define NN    49        // tokens per window
#define NP    64        // padded tokens per window
#define NWIN  64        // windows per image
#define BWIN  (BB*NWIN) // 2048
#define ROWS  (BWIN*NP) // 131072 padded rows
#define TOK   (BB*HH*WW_) // 100352
#define SCALE 0.17677669529663687f

// ---------------- workspace layout (bytes) ----------------
#define OFF_WQKV  ((size_t)0)
#define OFF_WPROJ (OFF_WQKV  + 221184)     // 576*192*2
#define OFF_WLIN  (OFF_WPROJ + 73728)      // 192*192*2
#define OFF_BIAS  (OFF_WLIN  + 73728)
#define OFF_XW    (OFF_BIAS  + 57856)      // 6*49*49*4 padded
#define OFF_Q     (OFF_XW    + 50331648)   // 2048*64*192*2
#define OFF_K     (OFF_Q     + 50331648)
#define OFF_V     (OFF_K     + 50331648)
#define OFF_CTX   (OFF_V     + 50331648)
#define OFF_XA    (OFF_CTX   + 50331648)

// ---------------- WMMA fragment helpers ----------------
__device__ __forceinline__ v8f wmma_f16(v16h a, v16h b, v8f c) {
  return __builtin_amdgcn_wmma_f32_16x16x32_f16(false, a, false, b, (short)0, c,
                                                false, false);
}

// A fragment 16x32 f16 from row-major LDS (row stride ld halfs), tile row tr.
// Lanes 0-15: M=lane, K {0..7,16..23}; lanes 16-31: M=lane-16, K {8..15,24..31}.
__device__ __forceinline__ v16h frag_a_lds(const _Float16* A, int tr, int ld) {
  int lane = threadIdx.x & 31;
  int m    = (lane & 15) + tr * 16;
  int koff = (lane >> 4) * 8;
  const _Float16* rowp = A + m * ld;
  v16h a;
#pragma unroll
  for (int j = 0; j < 8; ++j) {
    int ka = koff + ((j < 4) ? (2 * j) : (16 + 2 * (j - 4)));
    a[2 * j]     = rowp[ka];
    a[2 * j + 1] = rowp[ka + 1];
  }
  return a;
}

// B fragment 32x16 f16 from transposed LDS Bt[col][k] (col stride ld halfs),
// tile col tc. Lanes 0-15: N=lane, K {0..7,16..23}; lanes 16-31: N=lane-16, K {8..15,24..31}.
__device__ __forceinline__ v16h frag_b_lds(const _Float16* Bt, int tc, int ld) {
  int lane = threadIdx.x & 31;
  int n    = (lane & 15) + tc * 16;
  int koff = (lane >> 4) * 8;
  const _Float16* colp = Bt + n * ld;
  v16h b;
#pragma unroll
  for (int j = 0; j < 8; ++j) {
    int kb = koff + ((j < 4) ? (2 * j) : (16 + 2 * (j - 4)));
    b[2 * j]     = colp[kb];
    b[2 * j + 1] = colp[kb + 1];
  }
  return b;
}

// Shared 64x64 f16-GEMM inner loop: A[rowbase..+64][K], W[colbase..+64][K]
// (both row stride K = kdim), accumulates two 16x16 tiles per wave.
// As/Bs must each hold 2*64*32 halfs (double buffered in the async path).
__device__ __forceinline__ void gemm64x64(const _Float16* A, const _Float16* Wt,
                                          int rowbase, int colbase, int kdim,
                                          _Float16* As, _Float16* Bs,
                                          int tr, int tc0, v8f& acc0, v8f& acc1) {
  int t  = threadIdx.x;
  int rr = t >> 2;
  int p  = (t & 3) * 8;
  const _Float16* gA = A  + (size_t)(rowbase + rr) * kdim + p;
  const _Float16* gB = Wt + (size_t)(colbase + rr) * kdim + p;
  _Float16* sA = As + rr * 32 + p;
  _Float16* sB = Bs + rr * 32 + p;
#if USE_ASYNC_LDS
  // double-buffered async staging: copy for step kk+32 overlaps WMMAs of kk
  async_cp16(gA, sA);
  async_cp16(gB, sB);
  int buf = 0;
  for (int kk = 0; kk < kdim; kk += 32) {
    __builtin_amdgcn_s_wait_asynccnt(0);
    __syncthreads();                     // cur buffer ready for all waves
    if (kk + 32 < kdim) {
      int nb = buf ^ 1;
      async_cp16(gA + kk + 32, sA + nb * 2048);
      async_cp16(gB + kk + 32, sB + nb * 2048);
    }
    const _Float16* cA = As + buf * 2048;
    const _Float16* cB = Bs + buf * 2048;
    v16h a  = frag_a_lds(cA, tr, 32);
    v16h b0 = frag_b_lds(cB, tc0, 32);
    v16h b1 = frag_b_lds(cB, tc0 + 1, 32);
    acc0 = wmma_f16(a, b0, acc0);
    acc1 = wmma_f16(a, b1, acc1);
    buf ^= 1;
  }
#else
  for (int kk = 0; kk < kdim; kk += 32) {
    *(v8h*)sA = *(const v8h*)(gA + kk);
    *(v8h*)sB = *(const v8h*)(gB + kk);
    __syncthreads();
    v16h a  = frag_a_lds(As, tr, 32);
    v16h b0 = frag_b_lds(Bs, tc0, 32);
    v16h b1 = frag_b_lds(Bs, tc0 + 1, 32);
    acc0 = wmma_f16(a, b0, acc0);
    acc1 = wmma_f16(a, b1, acc1);
    __syncthreads();
  }
#endif
}

// ---------------- kernel 1: weights fp32 -> f16 ----------------
__global__ void k_convw(const float* qkv_w, const float* proj_w, const float* lin_w,
                        _Float16* wq, _Float16* wp, _Float16* wl) {
  int i = blockIdx.x * 256 + threadIdx.x;
  if (i < 576 * 192) wq[i] = (_Float16)qkv_w[i];
  if (i < 192 * 192) {
    wp[i] = (_Float16)proj_w[i];
    wl[i] = (_Float16)lin_w[i];
  }
}

// ---------------- kernel 2: rel-pos bias gather [6][49][49] ----------------
__global__ void k_bias(const float* table, const int* idx, float* bias6) {
  int i = blockIdx.x * 256 + threadIdx.x;
  if (i >= NH_ * NN * NN) return;
  int h  = i / (NN * NN);
  int ij = i % (NN * NN);
  bias6[i] = table[idx[ij] * NH_ + h];
}

// ---------------- kernel 3: LN + roll(-3) + window partition -> f16 --------
__global__ __launch_bounds__(64) void k_lnpart(const float* x, const float* g,
                                               const float* bt, _Float16* xw) {
  int blk = blockIdx.x;  // win*64 + n
  int win = blk >> 6;
  int n   = blk & 63;
  int t   = threadIdx.x;
  _Float16* dst = xw + (size_t)blk * CC;
  if (n >= NN) {
    dst[t] = (_Float16)0.0f; dst[t + 64] = (_Float16)0.0f; dst[t + 128] = (_Float16)0.0f;
    return;
  }
  int b = win >> 6, wio = (win >> 3) & 7, wjo = win & 7;
  int r = n / WS_, c = n % WS_;
  int hh = (wio * WS_ + r + SHIFT_) % HH;
  int ww = (wjo * WS_ + c + SHIFT_) % WW_;
  const float* src = x + ((size_t)(b * HH + hh) * WW_ + ww) * CC;
  float v0 = src[t], v1 = src[t + 64], v2 = src[t + 128];
  __shared__ float ssum[64], ssq[64];
  ssum[t] = v0 + v1 + v2;
  ssq[t]  = v0 * v0 + v1 * v1 + v2 * v2;
  __syncthreads();
#pragma unroll
  for (int s = 32; s > 0; s >>= 1) {
    if (t < s) { ssum[t] += ssum[t + s]; ssq[t] += ssq[t + s]; }
    __syncthreads();
  }
  float mu  = ssum[0] * (1.0f / CC);
  float var = ssq[0] * (1.0f / CC) - mu * mu;
  float inv = rsqrtf(var + 1e-5f);
  dst[t]       = (_Float16)((v0 - mu) * inv * g[t]       + bt[t]);
  dst[t + 64]  = (_Float16)((v1 - mu) * inv * g[t + 64]  + bt[t + 64]);
  dst[t + 128] = (_Float16)((v2 - mu) * inv * g[t + 128] + bt[t + 128]);
}

// ---------------- kernel 4: QKV GEMM (131072x192 @ 192x576) ----------------
__global__ __launch_bounds__(256) void k_qkv(const _Float16* xw, const _Float16* wq,
                                             const float* qkvb, _Float16* qh,
                                             _Float16* kh, _Float16* vh) {
  __shared__ __align__(16) _Float16 As[2 * 64 * 32];
  __shared__ __align__(16) _Float16 Bs[2 * 64 * 32];
  int rowbase = blockIdx.x * 64;
  int colbase = blockIdx.y * 64;
  int wv = threadIdx.x >> 5;
  int tr = wv & 3, tc0 = (wv >> 2) * 2;
  v8f a0 = {}, a1 = {};
  gemm64x64(xw, wq, rowbase, colbase, CC, As, Bs, tr, tc0, a0, a1);
  int lane = threadIdx.x & 31, n = lane & 15, hi = lane >> 4;
#pragma unroll
  for (int r = 0; r < 8; ++r) {
    size_t m = (size_t)rowbase + tr * 16 + r + 8 * hi;
#pragma unroll
    for (int u = 0; u < 2; ++u) {
      int oc = colbase + (tc0 + u) * 16 + n;
      float val = (u ? a1[r] : a0[r]) + qkvb[oc];
      if (oc < CC)          qh[m * CC + oc]            = (_Float16)(val * SCALE);
      else if (oc < 2 * CC) kh[m * CC + oc - CC]       = (_Float16)val;
      else                  vh[m * CC + oc - 2 * CC]   = (_Float16)val;
    }
  }
}

// ---------------- kernel 5: per-(window,head) attention ----------------
__global__ __launch_bounds__(128) void k_attn(const _Float16* qh, const _Float16* kh,
                                              const _Float16* vh, const float* bias6,
                                              const float* mask, _Float16* ctxh) {
  int win  = blockIdx.x;
  int head = blockIdx.y;
  __shared__ __align__(16) _Float16 qs[NP * HD_];   // [64][32]
  __shared__ __align__(16) _Float16 ks[NP * HD_];   // [64][32] (== K^T as Bt[col][k])
  __shared__ __align__(16) _Float16 vT[HD_ * NP];   // [32][64] v transposed
  __shared__ __align__(16) _Float16 ps[NP * NP];    // probs f16
  int t = threadIdx.x;
  const float* bh = bias6 + (size_t)head * NN * NN;
  const float* mh = mask + (size_t)(win & (NWIN - 1)) * NN * NN;
  __builtin_prefetch(bh + t * 16, 0, 0);            // global_prefetch for bias/mask
  __builtin_prefetch(mh + t * 16, 0, 0);
  if (t < 64) {
    const _Float16* s = qh + ((size_t)win * NP + t) * CC + head * HD_;
#pragma unroll
    for (int j = 0; j < 4; ++j) *(v8h*)(qs + t * HD_ + j * 8) = *(const v8h*)(s + j * 8);
  } else {
    int rr = t - 64;
    const _Float16* s = kh + ((size_t)win * NP + rr) * CC + head * HD_;
#pragma unroll
    for (int j = 0; j < 4; ++j) *(v8h*)(ks + rr * HD_ + j * 8) = *(const v8h*)(s + j * 8);
  }
  for (int e = t; e < NP * HD_; e += 128) {
    int j = e >> 5, d = e & 31;
    vT[d * NP + j] = vh[((size_t)win * NP + j) * CC + head * HD_ + d];
  }
  __syncthreads();

  int wv = t >> 5;                 // row tile
  v16h aq = frag_a_lds(qs, wv, HD_);
  v8f s0 = {}, s1 = {}, s2 = {}, s3 = {};
  s0 = wmma_f16(aq, frag_b_lds(ks, 0, HD_), s0);
  s1 = wmma_f16(aq, frag_b_lds(ks, 1, HD_), s1);
  s2 = wmma_f16(aq, frag_b_lds(ks, 2, HD_), s2);
  s3 = wmma_f16(aq, frag_b_lds(ks, 3, HD_), s3);

  int lane = t & 31, n = lane & 15, hi = lane >> 4;
#pragma unroll
  for (int r = 0; r < 8; ++r) {
    int i = wv * 16 + r + 8 * hi;
    float x[4] = {s0[r], s1[r], s2[r], s3[r]};
#pragma unroll
    for (int u = 0; u < 4; ++u) {
      int c = u * 16 + n;
      if (i < NN && c < NN) x[u] += bh[i * NN + c] + mh[i * NN + c];
      else                  x[u]  = -1e30f;
    }
    float mx = fmaxf(fmaxf(x[0], x[1]), fmaxf(x[2], x[3]));
#pragma unroll
    for (int o = 8; o >= 1; o >>= 1) mx = fmaxf(mx, __shfl_xor(mx, o, 32));
    float e0 = __expf(x[0] - mx), e1 = __expf(x[1] - mx);
    float e2 = __expf(x[2] - mx), e3 = __expf(x[3] - mx);
    float sm = e0 + e1 + e2 + e3;
#pragma unroll
    for (int o = 8; o >= 1; o >>= 1) sm += __shfl_xor(sm, o, 32);
    float inv = 1.0f / (sm + 1e-30f);
    ps[i * NP +      n] = (_Float16)(e0 * inv);
    ps[i * NP + 16 + n] = (_Float16)(e1 * inv);
    ps[i * NP + 32 + n] = (_Float16)(e2 * inv);
    ps[i * NP + 48 + n] = (_Float16)(e3 * inv);
  }
  __syncthreads();

  // ctx = probs @ v : [64x64] x [64x32] -> [64x32], 2 K-steps of 32
  v8f c0 = {}, c1 = {};
#pragma unroll
  for (int kstep = 0; kstep < 2; ++kstep) {
    v16h ap = frag_a_lds(ps + kstep * 32, wv, NP);
    c0 = wmma_f16(ap, frag_b_lds(vT + kstep * 32, 0, NP), c0);
    c1 = wmma_f16(ap, frag_b_lds(vT + kstep * 32, 1, NP), c1);
  }
#pragma unroll
  for (int r = 0; r < 8; ++r) {
    int i = wv * 16 + r + 8 * hi;
    ctxh[((size_t)win * NP + i) * CC + head * HD_ + n]      = (_Float16)c0[r];
    ctxh[((size_t)win * NP + i) * CC + head * HD_ + 16 + n] = (_Float16)c1[r];
  }
}

// ---------------- kernel 6: proj GEMM + window reverse + roll(+3) ----------
__global__ __launch_bounds__(256) void k_proj(const _Float16* ctxh, const _Float16* wp,
                                              const float* projb, _Float16* xa) {
  __shared__ __align__(16) _Float16 As[2 * 64 * 32];
  __shared__ __align__(16) _Float16 Bs[2 * 64 * 32];
  int rowbase = blockIdx.x * 64;
  int colbase = blockIdx.y * 64;
  int wv = threadIdx.x >> 5;
  int tr = wv & 3, tc0 = (wv >> 2) * 2;
  v8f a0 = {}, a1 = {};
  gemm64x64(ctxh, wp, rowbase, colbase, CC, As, Bs, tr, tc0, a0, a1);
  int lane = threadIdx.x & 31, n = lane & 15, hi = lane >> 4;
#pragma unroll
  for (int r = 0; r < 8; ++r) {
    size_t row = (size_t)rowbase + tr * 16 + r + 8 * hi;
    int win = (int)(row >> 6), nn = (int)(row & 63);
    if (nn >= NN) continue;
    int b = win >> 6, wio = (win >> 3) & 7, wjo = win & 7;
    int rr = nn / WS_, cc = nn % WS_;
    int hh = (wio * WS_ + rr + SHIFT_) % HH;
    int ww = (wjo * WS_ + cc + SHIFT_) % WW_;
    size_t tok = (size_t)(b * HH + hh) * WW_ + ww;
#pragma unroll
    for (int u = 0; u < 2; ++u) {
      int oc = colbase + (tc0 + u) * 16 + n;
      float val = (u ? a1[r] : a0[r]) + projb[oc];
      xa[tok * CC + oc] = (_Float16)val;
    }
  }
}

// ---------------- kernel 7: MLP GEMM + GELU + residual -> f32 out ----------
__global__ __launch_bounds__(256) void k_lin(const _Float16* xa, const _Float16* wl,
                                             const float* linb, const float* shortcut,
                                             float* out) {
  __shared__ __align__(16) _Float16 As[2 * 64 * 32];
  __shared__ __align__(16) _Float16 Bs[2 * 64 * 32];
  int rowbase = blockIdx.x * 64;
  int colbase = blockIdx.y * 64;
  int wv = threadIdx.x >> 5;
  int tr = wv & 3, tc0 = (wv >> 2) * 2;
  v8f a0 = {}, a1 = {};
  gemm64x64(xa, wl, rowbase, colbase, CC, As, Bs, tr, tc0, a0, a1);
  int lane = threadIdx.x & 31, n = lane & 15, hi = lane >> 4;
#pragma unroll
  for (int r = 0; r < 8; ++r) {
    size_t row = (size_t)rowbase + tr * 16 + r + 8 * hi;
#pragma unroll
    for (int u = 0; u < 2; ++u) {
      int oc = colbase + (tc0 + u) * 16 + n;
      float v = (u ? a1[r] : a0[r]) + linb[oc];
      float ge = 0.5f * v * (1.0f + erff(v * 0.70710678118654752f));
      out[row * CC + oc] = shortcut[row * CC + oc] + ge;
    }
  }
}

// ---------------- host launcher ----------------
extern "C" void kernel_launch(void* const* d_in, const int* in_sizes, int n_in,
                              void* d_out, int out_size, void* d_ws, size_t ws_size,
                              hipStream_t stream) {
  const float* inputs  = (const float*)d_in[0];
  const float* amask   = (const float*)d_in[1];
  const float* n1g     = (const float*)d_in[2];
  const float* n1b     = (const float*)d_in[3];
  const float* qkv_w   = (const float*)d_in[4];
  const float* qkv_b   = (const float*)d_in[5];
  const float* rp_tab  = (const float*)d_in[6];
  const int*   rp_idx  = (const int*)d_in[7];
  const float* proj_w  = (const float*)d_in[8];
  const float* proj_b  = (const float*)d_in[9];
  const float* lin_w   = (const float*)d_in[10];
  const float* lin_b   = (const float*)d_in[11];
  float* out = (float*)d_out;

  char* ws = (char*)d_ws;
  _Float16* wq   = (_Float16*)(ws + OFF_WQKV);
  _Float16* wp   = (_Float16*)(ws + OFF_WPROJ);
  _Float16* wl   = (_Float16*)(ws + OFF_WLIN);
  float*    bias = (float*)   (ws + OFF_BIAS);
  _Float16* xw   = (_Float16*)(ws + OFF_XW);
  _Float16* qh   = (_Float16*)(ws + OFF_Q);
  _Float16* kh   = (_Float16*)(ws + OFF_K);
  _Float16* vh   = (_Float16*)(ws + OFF_V);
  _Float16* ctxh = (_Float16*)(ws + OFF_CTX);
  _Float16* xa   = (_Float16*)(ws + OFF_XA);

  k_convw<<<(576 * 192 + 255) / 256, 256, 0, stream>>>(qkv_w, proj_w, lin_w, wq, wp, wl);
  k_bias<<<(NH_ * NN * NN + 255) / 256, 256, 0, stream>>>(rp_tab, rp_idx, bias);
  k_lnpart<<<BWIN * NP, 64, 0, stream>>>(inputs, n1g, n1b, xw);
  k_qkv<<<dim3(ROWS / 64, 576 / 64), 256, 0, stream>>>(xw, wq, qkv_b, qh, kh, vh);
  k_attn<<<dim3(BWIN, NH_), 128, 0, stream>>>(qh, kh, vh, bias, amask, ctxh);
  k_proj<<<dim3(ROWS / 64, CC / 64), 256, 0, stream>>>(ctxh, wp, proj_b, xa);
  k_lin<<<dim3(TOK / 64, CC / 64), 256, 0, stream>>>(xa, wl, lin_b, inputs, out);
}
